// MlpLSTM_28269474742888
// MI455X (gfx1250) — matrix operand order repaired and checked by reading, not compiled, not run on hardware
//
#include <hip/hip_runtime.h>
#include <hip/hip_bf16.h>
#include <stdint.h>

#define Bq 256
#define Sq 128
#define Fq 512
#define Hq 1024
#define Aq 18
#define BSq (Bq * Sq) // 32768

typedef __attribute__((ext_vector_type(16))) __bf16 v16bf;
typedef __attribute__((ext_vector_type(8)))  float  v8f;
typedef unsigned int u32x4 __attribute__((ext_vector_type(4)));
typedef int i32x8 __attribute__((ext_vector_type(8)));
typedef int i32x4 __attribute__((ext_vector_type(4)));

union FragU { v16bf v; uint4 q[2]; };

// A-matrix fragment (16x32 bf16): lane&15 = row M (baked into rowbase);
// lanes 0-15 hold K={0..7,16..23}, lanes 16-31 hold K={8..15,24..31}.
__device__ __forceinline__ v16bf ldfragA(const __bf16* rowbase, int kc, int lane) {
    int kh = (lane >> 4) << 3;
    const __bf16* p = rowbase + kc + kh;
    FragU f;
    f.q[0] = *(const uint4*)(p);
    f.q[1] = *(const uint4*)(p + 16);
    return f.v;
}

// B-matrix fragment (32x16 bf16): lane&15 = column N (row of row-major weight,
// baked into rowbase); lanes 0-15 hold K={0..7,16..23}, lanes 16-31 hold K={8..15,24..31}.
__device__ __forceinline__ v16bf ldfragB(const __bf16* rowbase, int kc, int lane) {
    int kh = (lane >> 4) << 3;
    const __bf16* p = rowbase + kc + kh;
    FragU f;
    f.q[0] = *(const uint4*)(p);
    f.q[1] = *(const uint4*)(p + 16);
    return f.v;
}

// A fragment from the padded LDS tile: logical row stride 2048B, +16B pad per
// 1024B segment (matches TDM pad_interval=7 / pad_amount=3) -> 2080B stride.
__device__ __forceinline__ v16bf ldfragA_lds(const char* sec, int row, int kc, int lane) {
    int kh = (lane >> 4) << 3;
    int k = kc + kh;
    int b0 = row * 2048 + k * 2;
    int b1 = b0 + 32;
    int o0 = b0 + ((b0 >> 10) << 4);
    int o1 = b1 + ((b1 >> 10) << 4);
    FragU f;
    f.q[0] = *(const uint4*)(sec + o0);
    f.q[1] = *(const uint4*)(sec + o1);
    return f.v;
}

__device__ __forceinline__ float sigmf(float x) { return 1.0f / (1.0f + __expf(-x)); }

// Build a 2-D TDM descriptor (D#) and issue tensor_load_to_lds (6-arg form).
// Tile: tile_d1 rows x tile_d0 elements (2-byte), row stride = stride_elems.
// LDS dest gets 16B padding per 1024B (pad_interval=7, pad_amount=3).
__device__ __forceinline__ void tdm_load_tile_2d(
    unsigned int lds_byte_addr, const void* gptr,
    unsigned int tensor_d0, unsigned int tensor_d1,
    unsigned int tile_d0, unsigned int tile_d1,
    unsigned long long stride_elems)
{
    unsigned long long ga = (unsigned long long)(uintptr_t)gptr;
    u32x4 g0;
    g0[0] = 1u;                                             // count=1, user mode
    g0[1] = lds_byte_addr;                                  // lds_addr
    g0[2] = (unsigned int)ga;                               // global_addr[31:0]
    g0[3] = (unsigned int)((ga >> 32) & 0x01FFFFFFu) | (2u << 30); // addr[56:32] | type=2
    i32x8 g1;
    g1[0] = (int)((1u << 16)            // data_size = 2 bytes
                | (1u << 20)            // pad_enable
                | (7u << 22)            // pad_interval: 256 dwords = 1024B
                | (3u << 25));          // pad_amount: 4 dwords = 16B
    g1[1] = (int)(tensor_d0 << 16);                         // tensor_dim0[15:0] @ bit48
    g1[2] = (int)((tensor_d0 >> 16) | (tensor_d1 << 16));   // dim0 hi | dim1 lo
    g1[3] = (int)((tensor_d1 >> 16) | (tile_d0 << 16));     // dim1 hi | tile_dim0
    g1[4] = (int)(tile_d1 & 0xFFFFu);                       // tile_dim1 (tile_dim2=0)
    g1[5] = (int)(unsigned int)(stride_elems & 0xFFFFFFFFu);// tensor_dim0_stride lo
    g1[6] = (int)(unsigned int)((stride_elems >> 32) & 0xFFFFu); // stride hi (dim1_stride=0)
    g1[7] = 0;
    i32x4 z4 = {0, 0, 0, 0};
    i32x8 z8 = {0, 0, 0, 0, 0, 0, 0, 0};
    __builtin_amdgcn_tensor_load_to_lds(g0, g1, z4, z4, z8, 0);
}

// ---------------- prep kernels ----------------
__global__ void cvt_bf16_kernel(const float* __restrict__ src, __bf16* __restrict__ dst, int n) {
    int i = blockIdx.x * blockDim.x + threadIdx.x;
    if (i < n) dst[i] = (__bf16)src[i];
}

__global__ void copy_f32_kernel(const float* __restrict__ src, float* __restrict__ dst, int n) {
    int i = blockIdx.x * blockDim.x + threadIdx.x;
    if (i < n) dst[i] = src[i];
}

// ---------------- MLP: x = relu(obs @ W1^T + b1), bf16 out ----------------
__global__ void __launch_bounds__(256) mlp_gemm_kernel(
    const __bf16* __restrict__ obs,  // [BS, F]
    const __bf16* __restrict__ W1,   // [H, F]
    const float*  __restrict__ b1,   // [H]
    __bf16* __restrict__ xout)       // [BS, H]
{
    int lane = threadIdx.x & 31;
    int wave = threadIdx.x >> 5;
    int ln   = lane & 15;
    int m0 = blockIdx.x * 16;
    int n0 = blockIdx.y * 512 + wave * 64;

    v8f acc[4] = {v8f{}, v8f{}, v8f{}, v8f{}};
    const __bf16* arow = obs + (size_t)(m0 + ln) * Fq;
    const __bf16* brow = W1 + (size_t)(n0 + ln) * Fq;

#pragma unroll
    for (int kc = 0; kc < Fq; kc += 32) {
        v16bf a = ldfragA(arow, kc, lane);
#pragma unroll
        for (int u = 0; u < 4; u++) {
            v16bf b = ldfragB(brow + (size_t)(u * 16) * Fq, kc, lane);
            acc[u] = __builtin_amdgcn_wmma_f32_16x16x32_bf16(
                false, a, false, b, (short)0, acc[u], false, false);
        }
    }

    int hi8 = (lane >> 4) << 3;
#pragma unroll
    for (int u = 0; u < 4; u++) {
        int col = n0 + u * 16 + ln;
        float bias = b1[col];
#pragma unroll
        for (int j = 0; j < 8; j++) {
            float v = acc[u][j] + bias;
            v = v > 0.0f ? v : 0.0f;
            xout[(size_t)(m0 + j + hi8) * Hq + col] = (__bf16)v;
        }
    }
}

// ---------------- LSTM step ----------------
// Block: 32 batch rows x 128 h cols. TDM stages the block's A-tile
// (x_t rows + h_prev rows) into LDS once; each wave holds 8 accumulators
// (2 batch tiles x 4 gates) and double-buffers B fragments in registers.
#define LDS_SEC 66560 // 32 rows * 2080 bytes (2048 + 2x16B pad)

__global__ void __launch_bounds__(256) lstm_step_kernel(
    const __bf16* __restrict__ xbf,    // [BS, H]
    const __bf16* __restrict__ hprev,  // [B, H]
    const __bf16* __restrict__ Wih,    // [4H, H]
    const __bf16* __restrict__ Whh,    // [4H, H]
    const float* __restrict__ b_ih,    // [4H]
    const float* __restrict__ b_hh,    // [4H]
    float*  __restrict__ c,            // [B, H] in/out
    __bf16* __restrict__ hnext,        // [B, H]
    __bf16* __restrict__ hs,           // [BS, H]
    int t)
{
    __shared__ __align__(16) char smem[2 * LDS_SEC];
    int lane = threadIdx.x & 31;
    int wave = threadIdx.x >> 5;
    int ln   = lane & 15;
    int m0 = blockIdx.y * 32;                 // batch tile base (2 x 16)
    int h0 = blockIdx.x * 128 + wave * 16;    // hidden-col tile base

    // Wave 0 DMAs the A-tile into LDS via the Tensor Data Mover.
    if (threadIdx.x < 32) {
        unsigned int lds0 = (unsigned int)(uintptr_t)(void*)&smem[0];
        tdm_load_tile_2d(lds0, xbf + ((size_t)m0 * Sq + t) * Hq,
                         Hq, 32u, Hq, 32u, (unsigned long long)Sq * Hq);
        tdm_load_tile_2d(lds0 + LDS_SEC, hprev + (size_t)m0 * Hq,
                         Hq, 32u, Hq, 32u, (unsigned long long)Hq);
        __builtin_amdgcn_s_wait_tensorcnt(0);
    }
    __syncthreads();

    v8f acc[8];
#pragma unroll
    for (int i = 0; i < 8; i++) acc[i] = v8f{};

    const __bf16* bw_ih = Wih + (size_t)(h0 + ln) * Hq;
    const __bf16* bw_hh = Whh + (size_t)(h0 + ln) * Hq;
    const char* xsec = smem;
    const char* hsec = smem + LDS_SEC;

    v16bf bbuf[2][4];
#pragma unroll
    for (int g = 0; g < 4; g++)
        bbuf[0][g] = ldfragB(bw_ih + (size_t)g * Hq * Hq, 0, lane);

    // K = 0..1023 : x_t @ W_ih^T
#pragma unroll
    for (int kc = 0; kc < Hq; kc += 32) {
        int cur = (kc >> 5) & 1, nxt = cur ^ 1;
        if (kc + 32 < Hq) {
#pragma unroll
            for (int g = 0; g < 4; g++)
                bbuf[nxt][g] = ldfragB(bw_ih + (size_t)g * Hq * Hq, kc + 32, lane);
        } else {
#pragma unroll
            for (int g = 0; g < 4; g++)
                bbuf[nxt][g] = ldfragB(bw_hh + (size_t)g * Hq * Hq, 0, lane);
        }
        v16bf a0 = ldfragA_lds(xsec, ln, kc, lane);
        v16bf a1 = ldfragA_lds(xsec, 16 + ln, kc, lane);
#pragma unroll
        for (int g = 0; g < 4; g++) {
            acc[g]     = __builtin_amdgcn_wmma_f32_16x16x32_bf16(
                false, a0, false, bbuf[cur][g], (short)0, acc[g], false, false);
            acc[4 + g] = __builtin_amdgcn_wmma_f32_16x16x32_bf16(
                false, a1, false, bbuf[cur][g], (short)0, acc[4 + g], false, false);
        }
    }
    // K = 1024..2047 : h_{t-1} @ W_hh^T (parity continues: bbuf[0] holds chunk 0)
#pragma unroll
    for (int kc = 0; kc < Hq; kc += 32) {
        int cur = (kc >> 5) & 1, nxt = cur ^ 1;
        if (kc + 32 < Hq) {
#pragma unroll
            for (int g = 0; g < 4; g++)
                bbuf[nxt][g] = ldfragB(bw_hh + (size_t)g * Hq * Hq, kc + 32, lane);
        }
        v16bf a0 = ldfragA_lds(hsec, ln, kc, lane);
        v16bf a1 = ldfragA_lds(hsec, 16 + ln, kc, lane);
#pragma unroll
        for (int g = 0; g < 4; g++) {
            acc[g]     = __builtin_amdgcn_wmma_f32_16x16x32_bf16(
                false, a0, false, bbuf[cur][g], (short)0, acc[g], false, false);
            acc[4 + g] = __builtin_amdgcn_wmma_f32_16x16x32_bf16(
                false, a1, false, bbuf[cur][g], (short)0, acc[4 + g], false, false);
        }
    }

    int h = h0 + ln;
    float bi = b_ih[h]          + b_hh[h];
    float bf = b_ih[Hq + h]     + b_hh[Hq + h];
    float bg = b_ih[2 * Hq + h] + b_hh[2 * Hq + h];
    float bo = b_ih[3 * Hq + h] + b_hh[3 * Hq + h];
    int hi8 = (lane >> 4) << 3;
#pragma unroll
    for (int mt = 0; mt < 2; mt++) {
#pragma unroll
        for (int j = 0; j < 8; j++) {
            int bb = m0 + mt * 16 + j + hi8;
            size_t ci = (size_t)bb * Hq + h;
            float iv = sigmf(acc[mt * 4 + 0][j] + bi);
            float fv = sigmf(acc[mt * 4 + 1][j] + bf);
            float gv = tanhf(acc[mt * 4 + 2][j] + bg);
            float ov = sigmf(acc[mt * 4 + 3][j] + bo);
            float cn = fv * c[ci] + iv * gv;
            c[ci] = cn;
            float hn = ov * tanhf(cn);
            __bf16 hb = (__bf16)hn;
            hnext[ci] = hb;
            hs[((size_t)bb * Sq + t) * Hq + h] = hb;
        }
    }
}

// ---------------- heads: logits/value + log_softmax/entropy ----------------
__global__ void __launch_bounds__(256) heads_kernel(
    const __bf16* __restrict__ hs,  // [BS, H]
    const __bf16* __restrict__ Wl,  // [A, H]
    const __bf16* __restrict__ Wv,  // [H]
    const float* __restrict__ b_l,  // [A]
    const float* __restrict__ b_v,  // [1]
    const int* __restrict__ acts,   // [BS]
    float* __restrict__ out)        // [3*BS]: logp | entropy | value
{
    int lane = threadIdx.x & 31;
    int row = blockIdx.x * 8 + (threadIdx.x >> 5);
    const __bf16* hr = hs + (size_t)row * Hq;

    float acc[Aq];
#pragma unroll
    for (int a = 0; a < Aq; a++) acc[a] = 0.0f;
    float vacc = 0.0f;

    for (int hh = lane; hh < Hq; hh += 32) {
        float hv = (float)hr[hh];
        vacc += hv * (float)Wv[hh];
#pragma unroll
        for (int a = 0; a < Aq; a++) acc[a] += hv * (float)Wl[a * Hq + hh];
    }
#pragma unroll
    for (int a = 0; a < Aq; a++)
#pragma unroll
        for (int off = 16; off > 0; off >>= 1) acc[a] += __shfl_xor(acc[a], off, 32);
    for (int off = 16; off > 0; off >>= 1) vacc += __shfl_xor(vacc, off, 32);

    if (lane == 0) {
        float z[Aq];
        float m = -3.4e38f;
#pragma unroll
        for (int a = 0; a < Aq; a++) { z[a] = acc[a] + b_l[a]; m = fmaxf(m, z[a]); }
        float se = 0.0f;
#pragma unroll
        for (int a = 0; a < Aq; a++) se += __expf(z[a] - m);
        float lse = m + __logf(se);
        float inv_se = 1.0f / se;
        float ent = 0.0f;
#pragma unroll
        for (int a = 0; a < Aq; a++) {
            float p = __expf(z[a] - m) * inv_se;
            ent += p * (lse - z[a]);
        }
        int act = acts[row];
        out[row] = z[act] - lse;
        out[BSq + row] = ent;
        out[2 * BSq + row] = vacc + b_v[0];
    }
}

extern "C" void kernel_launch(void* const* d_in, const int* in_sizes, int n_in,
                              void* d_out, int out_size, void* d_ws, size_t ws_size,
                              hipStream_t stream) {
    (void)in_sizes; (void)n_in; (void)out_size; (void)ws_size;
    const float* obs = (const float*)d_in[0];
    const float* hx  = (const float*)d_in[1];
    const float* cx  = (const float*)d_in[2];
    const int*   bacts = (const int*)d_in[3];
    const float* W1  = (const float*)d_in[4];
    const float* b1  = (const float*)d_in[5];
    const float* Wih = (const float*)d_in[6];
    const float* bih = (const float*)d_in[7];
    const float* Whh = (const float*)d_in[8];
    const float* bhh = (const float*)d_in[9];
    const float* Wv  = (const float*)d_in[10];
    const float* bv  = (const float*)d_in[11];
    const float* Wl  = (const float*)d_in[12];
    const float* bl  = (const float*)d_in[13];

    char* ws = (char*)d_ws;
    size_t off = 0;
    auto carve = [&](size_t bytes) -> char* {
        char* p = ws + off;
        off = (off + bytes + 255) & ~(size_t)255;
        return p;
    };
    __bf16* obs_bf = (__bf16*)carve((size_t)BSq * Fq * 2);
    __bf16* W1_bf  = (__bf16*)carve((size_t)Hq * Fq * 2);
    __bf16* Wih_bf = (__bf16*)carve((size_t)4 * Hq * Hq * 2);
    __bf16* Whh_bf = (__bf16*)carve((size_t)4 * Hq * Hq * 2);
    __bf16* Wl_bf  = (__bf16*)carve((size_t)Aq * Hq * 2);
    __bf16* Wv_bf  = (__bf16*)carve((size_t)Hq * 2);
    __bf16* x_bf   = (__bf16*)carve((size_t)BSq * Hq * 2);
    __bf16* hs_bf  = (__bf16*)carve((size_t)BSq * Hq * 2);
    __bf16* h0_bf  = (__bf16*)carve((size_t)Bq * Hq * 2);
    __bf16* h1_bf  = (__bf16*)carve((size_t)Bq * Hq * 2);
    float*  c_ws   = (float*)carve((size_t)Bq * Hq * 4);

    auto cvt = [&](const float* s, __bf16* d, int n) {
        cvt_bf16_kernel<<<(n + 255) / 256, 256, 0, stream>>>(s, d, n);
    };
    cvt(obs, obs_bf, BSq * Fq);
    cvt(W1, W1_bf, Hq * Fq);
    cvt(Wih, Wih_bf, 4 * Hq * Hq);
    cvt(Whh, Whh_bf, 4 * Hq * Hq);
    cvt(Wl, Wl_bf, Aq * Hq);
    cvt(Wv, Wv_bf, Hq);
    cvt(hx, h0_bf, Bq * Hq);
    copy_f32_kernel<<<(Bq * Hq + 255) / 256, 256, 0, stream>>>(cx, c_ws, Bq * Hq);

    dim3 gm(BSq / 16, Hq / 512);
    mlp_gemm_kernel<<<gm, 256, 0, stream>>>(obs_bf, W1_bf, b1, x_bf);

    dim3 gs(Hq / 128, Bq / 32); // 8 x 8 blocks, 8 waves/block
    for (int t = 0; t < Sq; t++) {
        const __bf16* hp = (t & 1) ? h1_bf : h0_bf;
        __bf16* hn = (t & 1) ? h0_bf : h1_bf;
        lstm_step_kernel<<<gs, 256, 0, stream>>>(x_bf, hp, Wih_bf, Whh_bf,
                                                 bih, bhh, c_ws, hn, hs_bf, t);
    }

    heads_kernel<<<BSq / 8, 256, 0, stream>>>(hs_bf, Wl_bf, Wv_bf, bl, bv,
                                              bacts, (float*)d_out);
}
